// LSTM_26276609917602
// MI455X (gfx1250) — matrix operand order repaired and checked by reading, not compiled, not run on hardware
//
#include <hip/hip_runtime.h>
#include <hip/hip_bf16.h>

// ---------------------------------------------------------------------------
// LSTM LM forward for MI455X (gfx1250, wave32, WMMA).
// B=16, T=256, D=256, H=512, V=32000.
// ---------------------------------------------------------------------------

typedef __attribute__((ext_vector_type(16))) _Float16 v16h;
typedef __attribute__((ext_vector_type(8)))  _Float16 v8h;
typedef __attribute__((ext_vector_type(8)))  float    v8f;
typedef int v4i_gcc __attribute__((vector_size(16)));   // matches builtin param

#define LSTM_B 16
#define LSTM_T 256
#define LSTM_D 256
#define LSTM_H 512
#define LSTM_G 2048   // 4*H
#define LSTM_V 32000

// ---- optional async global->LDS path (guarded: falls back to sync copy) ----
#if defined(__has_builtin)
#  if __has_builtin(__builtin_amdgcn_global_load_async_to_lds_b128) && \
      __has_builtin(__builtin_amdgcn_s_wait_asynccnt)
#    define USE_ASYNC_LDS 1
#  endif
#endif

// ---------------------------------------------------------------------------
// WMMA helpers (CDNA5 layouts, wave32)
// ---------------------------------------------------------------------------
__device__ __forceinline__ v8f wmma_f16(v16h a, v16h b, v8f c) {
  // D(16x16,f32) = A(16x32,f16) * B(32x16,f16) + C
  return __builtin_amdgcn_wmma_f32_16x16x32_f16(
      /*neg_a=*/false, a, /*neg_b=*/false, b,
      /*c_mod=*/(short)0, c, /*reuse_a=*/false, /*reuse_b=*/false);
}

// A fragment, 16x32 f16, from row-major (rows x ld) matrix at (row0=0, k0=0 of base).
// Lane l (<16): row=l, K = [0..15].
// Lane l (>=16): row=l-16, K = [16..31].
__device__ __forceinline__ v16h load_A(const _Float16* base, int ld, unsigned lane) {
  const int r     = lane & 15;
  const int khalf = (lane < 16) ? 0 : 16;
  const _Float16* p = base + r * ld + khalf;
  v8h lo = *(const v8h*)(p);        // K = khalf .. khalf+7
  v8h hi = *(const v8h*)(p + 8);    // K = khalf+8 .. khalf+15
  v16h a;
#pragma unroll
  for (int i = 0; i < 8; ++i) { a[i] = lo[i]; a[8 + i] = hi[i]; }
  return a;
}

// B fragment, 32x16 f16, where B[k][n] = W[n][k], W row-major with stride ld.
// Lane l: n = l&15, K = (l<16 ? 0..15 : 16..31) -> 16 contiguous f16 from W's row.
__device__ __forceinline__ v16h load_B(const _Float16* w, int ld, unsigned lane) {
  const int n  = lane & 15;
  const int kb = (lane < 16) ? 0 : 16;
  return *(const v16h*)(w + n * ld + kb);
}

__device__ __forceinline__ float sigmoidf_(float x) {
  return 1.0f / (1.0f + __expf(-x));
}

// ---------------------------------------------------------------------------
// Kernel A: fp32 -> fp16 weight conversion (grid-stride)
// ---------------------------------------------------------------------------
__global__ void cvt_f32_to_f16(const float* __restrict__ src,
                               _Float16* __restrict__ dst, int n) {
  for (int i = blockIdx.x * blockDim.x + threadIdx.x; i < n;
       i += gridDim.x * blockDim.x)
    dst[i] = (_Float16)src[i];
}

// ---------------------------------------------------------------------------
// Kernel B: gather x_seq + x_gates = x @ W_ih^T + b_ih + b_hh   (fp32 out)
//   grid = (T, 4), block = 256 (8 waves). Block -> one timestep (16 rows),
//   one quarter of the 2048 gate columns. Wave -> 4 N-tiles.
// ---------------------------------------------------------------------------
__global__ void xgates_kernel(const float* __restrict__ images,
                              const int* __restrict__ labels,
                              const float* __restrict__ embed,
                              const _Float16* __restrict__ wih16,
                              const float* __restrict__ b_ih,
                              const float* __restrict__ b_hh,
                              float* __restrict__ xg) {
  __shared__ __align__(32) _Float16 xrow[LSTM_B * LSTM_D];  // 16 x 256 f16 = 8 KB

  const int t = blockIdx.x;

  // Stage gathered input rows (convert to f16). thread -> (row, 16 cols)
  {
    const int r  = threadIdx.x >> 4;
    const int c0 = (threadIdx.x & 15) * 16;
    const float* srcp;
    if (t == 0) {
      srcp = images + r * LSTM_D;
    } else {
      const int lab = labels[t * LSTM_B + r];
      srcp = embed + (size_t)lab * LSTM_D;
    }
#pragma unroll
    for (int i = 0; i < 16; ++i)
      xrow[r * LSTM_D + c0 + i] = (_Float16)srcp[c0 + i];
  }
  __syncthreads();

  const unsigned lane = threadIdx.x & 31;
  const unsigned w    = threadIdx.x >> 5;  // 0..7

  v8f acc[4];
#pragma unroll
  for (int j = 0; j < 4; ++j) acc[j] = (v8f)0.0f;

#pragma unroll
  for (int ch = 0; ch < LSTM_D / 32; ++ch) {  // 8 K-chunks
    const v16h a = load_A(xrow + ch * 32, LSTM_D, lane);
#pragma unroll
    for (int j = 0; j < 4; ++j) {
      const int nt = blockIdx.y * 32 + w * 4 + j;  // 0..127
      const v16h b =
          load_B(wih16 + (size_t)nt * 16 * LSTM_D + ch * 32, LSTM_D, lane);
      acc[j] = wmma_f16(a, b, acc[j]);
    }
  }

  // bias + store (D layout: m = v + 8*(lane>=16), n = lane&15)
  const int mo = (lane < 16) ? 0 : 8;
#pragma unroll
  for (int j = 0; j < 4; ++j) {
    const int nt  = blockIdx.y * 32 + w * 4 + j;
    const int col = nt * 16 + (lane & 15);
    const float bias = b_ih[col] + b_hh[col];
#pragma unroll
    for (int v = 0; v < 8; ++v) {
      const int m = v + mo;
      xg[((size_t)t * LSTM_B + m) * LSTM_G + col] = acc[j][v] + bias;
    }
  }
}

// ---------------------------------------------------------------------------
// Kernel C: recurrent LSTM scan. ONE workgroup, 1024 threads = 32 waves.
//   Wave w owns h-columns [w*16, w*16+16): computes i/f/g/o gate tiles,
//   keeps c in registers, h round-trips through LDS (f16) each step.
// ---------------------------------------------------------------------------
__global__ void __launch_bounds__(1024)
lstm_scan_kernel(const float* __restrict__ xg,
                 const _Float16* __restrict__ whh16,
                 _Float16* __restrict__ hs16) {
  __shared__ __align__(32) _Float16 hbuf[LSTM_B * LSTM_H];  // 16 x 512 f16 = 16 KB

  // zero-init h
  for (int i = threadIdx.x; i < LSTM_B * LSTM_H; i += blockDim.x)
    hbuf[i] = (_Float16)0.0f;
  __syncthreads();

  const unsigned lane = threadIdx.x & 31;
  const unsigned w    = threadIdx.x >> 5;  // 0..31: h-tile index
  const int mo  = (lane < 16) ? 0 : 8;
  const int n   = lane & 15;
  const int col = w * 16 + n;  // h column this lane produces

  v8f c_state = (v8f)0.0f;

  for (int t = 0; t < LSTM_T; ++t) {
    const float* xgrow = xg + (size_t)t * LSTM_B * LSTM_G;

    // prefetch next timestep's x_gates slice (non-blocking, cacheline pull)
    if (t + 1 < LSTM_T)
      __builtin_prefetch(xg + (size_t)(t + 1) * LSTM_B * LSTM_G +
                             (size_t)mo * LSTM_G + col, 0, 1);

    v8f gi = (v8f)0.0f, gf = (v8f)0.0f, gg = (v8f)0.0f, go = (v8f)0.0f;

#pragma unroll
    for (int ch = 0; ch < LSTM_H / 32; ++ch) {  // 16 K-chunks over H
      const v16h a = load_A(hbuf + ch * 32, LSTM_H, lane);
      gi = wmma_f16(a, load_B(whh16 + ((size_t)(0 * LSTM_H + w * 16)) * LSTM_H + ch * 32, LSTM_H, lane), gi);
      gf = wmma_f16(a, load_B(whh16 + ((size_t)(1 * LSTM_H + w * 16)) * LSTM_H + ch * 32, LSTM_H, lane), gf);
      gg = wmma_f16(a, load_B(whh16 + ((size_t)(2 * LSTM_H + w * 16)) * LSTM_H + ch * 32, LSTM_H, lane), gg);
      go = wmma_f16(a, load_B(whh16 + ((size_t)(3 * LSTM_H + w * 16)) * LSTM_H + ch * 32, LSTM_H, lane), go);
    }

    float hval[8];
#pragma unroll
    for (int v = 0; v < 8; ++v) {
      const int m = v + mo;
      const float iv = sigmoidf_(gi[v] + xgrow[(size_t)m * LSTM_G + 0 * LSTM_H + col]);
      const float fv = sigmoidf_(gf[v] + xgrow[(size_t)m * LSTM_G + 1 * LSTM_H + col]);
      const float gv = tanhf    (gg[v] + xgrow[(size_t)m * LSTM_G + 2 * LSTM_H + col]);
      const float ov = sigmoidf_(go[v] + xgrow[(size_t)m * LSTM_G + 3 * LSTM_H + col]);
      c_state[v] = fv * c_state[v] + iv * gv;
      hval[v]    = ov * tanhf(c_state[v]);
    }

    __syncthreads();  // everyone done reading old h
#pragma unroll
    for (int v = 0; v < 8; ++v) {
      const int m = v + mo;
      const _Float16 hh = (_Float16)hval[v];
      hbuf[m * LSTM_H + col] = hh;
      hs16[((size_t)t * LSTM_B + m) * LSTM_H + col] = hh;
    }
    __syncthreads();  // new h visible for next step
  }
}

// ---------------------------------------------------------------------------
// Kernel D: out = hs @ W_fc^T + b_fc      (M=4096, N=32000, K=512)
//   grid = (256 M-tiles, 63), block = 256 (8 waves). Block -> 16 rows x 512
//   cols. A-tile (16x512 f16) staged in LDS via async global->LDS loads.
// ---------------------------------------------------------------------------
__global__ void fc_kernel(const _Float16* __restrict__ hs16,
                          const _Float16* __restrict__ wfc16,
                          const float* __restrict__ b_fc,
                          float* __restrict__ out) {
  __shared__ __align__(32) _Float16 hrow[LSTM_B * LSTM_H];  // 16 x 512 f16 = 16 KB

  const _Float16* src = hs16 + (size_t)blockIdx.x * LSTM_B * LSTM_H;

#if USE_ASYNC_LDS
#pragma unroll
  for (int i = 0; i < 4; ++i) {
    const int off = (threadIdx.x + i * 256) * 8;  // 16 bytes per thread/pass
    __builtin_amdgcn_global_load_async_to_lds_b128(
        (__attribute__((address_space(1))) v4i_gcc*)(src + off),
        (__attribute__((address_space(3))) v4i_gcc*)(&hrow[off]),
        0, 0);
  }
  __builtin_amdgcn_s_wait_asynccnt(0);
  __syncthreads();
#else
#pragma unroll
  for (int i = 0; i < 4; ++i) {
    const int off = (threadIdx.x + i * 256) * 8;
    *(v8h*)(&hrow[off]) = *(const v8h*)(src + off);
  }
  __syncthreads();
#endif

  const unsigned lane = threadIdx.x & 31;
  const unsigned w    = threadIdx.x >> 5;  // 0..7
  const int mo = (lane < 16) ? 0 : 8;
  const int m0 = blockIdx.x * 16;

#pragma unroll
  for (int j = 0; j < 4; ++j) {
    const int nt = blockIdx.y * 32 + w * 4 + j;  // N-tile (16 vocab cols)
    if (nt >= LSTM_V / 16) continue;             // wave-uniform guard

    v8f acc = (v8f)0.0f;
    const _Float16* wbase = wfc16 + (size_t)nt * 16 * LSTM_H;
#pragma unroll
    for (int ch = 0; ch < LSTM_H / 32; ++ch) {  // 16 K-chunks
      const v16h a = load_A(hrow + ch * 32, LSTM_H, lane);
      const v16h b = load_B(wbase + ch * 32, LSTM_H, lane);
      acc = wmma_f16(a, b, acc);
    }

    const int col   = nt * 16 + (lane & 15);
    const float bias = b_fc[col];
#pragma unroll
    for (int v = 0; v < 8; ++v) {
      const int m = v + mo;
      out[((size_t)(m0 + m)) * LSTM_V + col] = acc[v] + bias;
    }
  }
}

// ---------------------------------------------------------------------------
// Host launcher
// ---------------------------------------------------------------------------
extern "C" void kernel_launch(void* const* d_in, const int* in_sizes, int n_in,
                              void* d_out, int out_size, void* d_ws, size_t ws_size,
                              hipStream_t stream) {
  (void)in_sizes; (void)n_in; (void)out_size; (void)ws_size;

  const float* images = (const float*)d_in[0];
  const int*   labels = (const int*)d_in[1];
  const float* embed  = (const float*)d_in[2];
  const float* W_ih   = (const float*)d_in[3];
  const float* W_hh   = (const float*)d_in[4];
  const float* b_ih   = (const float*)d_in[5];
  const float* b_hh   = (const float*)d_in[6];
  const float* W_fc   = (const float*)d_in[7];
  const float* b_fc   = (const float*)d_in[8];
  float* out = (float*)d_out;

  // Workspace layout (bytes)
  char* ws = (char*)d_ws;
  _Float16* wih16 = (_Float16*)(ws);                       // 2048*256  f16 =  1.0 MB
  _Float16* whh16 = (_Float16*)(ws + 1048576);             // 2048*512  f16 =  2.0 MB
  _Float16* wfc16 = (_Float16*)(ws + 3145728);             // 32000*512 f16 = 32.8 MB
  float*    xg    = (float*)   (ws + 35913728);            // 4096*2048 f32 = 33.6 MB
  _Float16* hs16  = (_Float16*)(ws + 69468160);            // 4096*512  f16 =  4.2 MB

  // A: weight conversion
  cvt_f32_to_f16<<<1024, 256, 0, stream>>>(W_ih, wih16, LSTM_G * LSTM_D);
  cvt_f32_to_f16<<<1024, 256, 0, stream>>>(W_hh, whh16, LSTM_G * LSTM_H);
  cvt_f32_to_f16<<<4096, 256, 0, stream>>>(W_fc, wfc16, LSTM_V * LSTM_H);

  // B: embedding gather + input-side gate GEMM (+ both biases)
  xgates_kernel<<<dim3(LSTM_T, 4), 256, 0, stream>>>(
      images, labels, embed, wih16, b_ih, b_hh, xg);

  // C: sequential LSTM scan (single persistent workgroup, 32 waves)
  lstm_scan_kernel<<<1, 1024, 0, stream>>>(xg, whh16, hs16);

  // D: vocab projection
  fc_kernel<<<dim3(LSTM_T * LSTM_B / 16, 63), 256, 0, stream>>>(
      hs16, wfc16, b_fc, out);
}
